// T6Attention_40183714021950
// MI455X (gfx1250) — matrix operand order repaired and checked by reading, not compiled, not run on hardware
//
#include <hip/hip_runtime.h>
#include <hip/hip_bf16.h>
#include <math.h>

// ---------------- problem constants ----------------
constexpr int Bn = 2;
constexpr int Sn = 2048;
constexpr int Dm = 1024;
constexpr int Hn = 16;
constexpr int DK = 64;
constexpr int QR = 6;
constexpr int RK = 2;

constexpr int NAq = Hn * QR;   // 96
constexpr int NAk = Hn * RK;   // 32
constexpr int NAv = Hn * RK;   // 32
constexpr int NBq = QR * DK;   // 384
constexpr int NBk = RK * DK;   // 128
constexpr int NBv = RK * DK;   // 128

typedef __attribute__((ext_vector_type(16))) __bf16 v16bf;
typedef __attribute__((ext_vector_type(8)))  __bf16 v8bf;
typedef __attribute__((ext_vector_type(8)))  float  v8f;

#define WMMA_BF16(a, b, c) \
  __builtin_amdgcn_wmma_f32_16x16x32_bf16(false, (a), false, (b), (short)0, (c), false, false)

// A-fragment (16x32 bf16, MxK) load. `base` = &A[row m][k0] of a row-major
// matrix. Contiguous order (matches B): lanes 0-15 (half=0) hold K 0-15;
// lanes 16-31 (half=1) hold K 16-31. One 32B load per lane.
__device__ inline v16bf load_a_frag(const __bf16* base, int half) {
  return *(const v16bf*)(base + half * 16);
}

// ---------------- fp32 -> bf16 convert ----------------
__global__ void k_cvt_bf16(const float* __restrict__ src, __bf16* __restrict__ dst, int n) {
  int i = blockIdx.x * blockDim.x + threadIdx.x;
  if (i < n) dst[i] = (__bf16)src[i];
}

// fp32[K,N] -> bf16 transposed [N,K] (for WMMA B operands: K contiguous per lane)
__global__ void k_cvt_t_bf16(const float* __restrict__ src, __bf16* __restrict__ dst,
                             int K, int N) {
  int i = blockIdx.x * blockDim.x + threadIdx.x;  // index into dst [N,K]
  if (i < K * N) {
    int n = i / K, k = i % K;
    dst[i] = (__bf16)src[(size_t)k * N + n];
  }
}

// ---------------- bf16 WMMA GEMM: C[M,N] f32 = A[M,K] * Bt[N,K]^T ----------------
// block = 256 threads = 8 waves (2x4); each wave computes a 32x32 C patch
// (2 M-tiles x 2 N-tiles, 4 WMMAs per K-step, software-pipelined loads).
// Requires: M multiple of 32, N multiple of 32, K multiple of 32.
__global__ void __launch_bounds__(256)
k_gemm_bf16(const __bf16* __restrict__ A, const __bf16* __restrict__ Bt,
            float* __restrict__ C, int M, int N, int K) {
  const int lane = threadIdx.x & 31;
  const int wid  = threadIdx.x >> 5;
  const int tM0 = (blockIdx.y * 2 + (wid >> 2)) * 2;  // pair of 16-row tiles
  const int tN0 = (blockIdx.x * 4 + (wid & 3)) * 2;   // pair of 16-col tiles
  if (tM0 * 16 >= M || tN0 * 16 >= N) return;  // full 32x32 patch valid otherwise

  const int nn   = lane & 15;
  const int half = lane >> 4;
  const __bf16* arow0 = A  + (size_t)(tM0 * 16 + nn) * K;
  const __bf16* arow1 = arow0 + (size_t)16 * K;
  const __bf16* brow0 = Bt + (size_t)(tN0 * 16 + nn) * K;
  const __bf16* brow1 = brow0 + (size_t)16 * K;

  v8f a00 = {}, a01 = {}, a10 = {}, a11 = {};

  // stage 0 of the pipeline
  v16bf af0 = load_a_frag(arow0, half);
  v16bf af1 = load_a_frag(arow1, half);
  v16bf bf0 = *(const v16bf*)(brow0 + half * 16);
  v16bf bf1 = *(const v16bf*)(brow1 + half * 16);

  for (int k0 = 0; k0 < K; k0 += 32) {
    // issue next-step loads before consuming current fragments
    const int kn = (k0 + 32 < K) ? k0 + 32 : k0;       // clamped (branch-free)
    const int kp = (kn + 32 < K) ? kn + 32 : kn;
    __builtin_prefetch(arow0 + kp, 0, 1);              // global_prefetch_b8
    __builtin_prefetch(brow0 + kp, 0, 1);
    v16bf naf0 = load_a_frag(arow0 + kn, half);
    v16bf naf1 = load_a_frag(arow1 + kn, half);
    v16bf nbf0 = *(const v16bf*)(brow0 + kn + half * 16);
    v16bf nbf1 = *(const v16bf*)(brow1 + kn + half * 16);

    a00 = WMMA_BF16(af0, bf0, a00);
    a01 = WMMA_BF16(af0, bf1, a01);
    a10 = WMMA_BF16(af1, bf0, a10);
    a11 = WMMA_BF16(af1, bf1, a11);

    af0 = naf0; af1 = naf1; bf0 = nbf0; bf1 = nbf1;
  }

  // C tiles: VGPR j holds row (tile*16 + j + 8*half), column nn within tile.
  const int n0 = tN0 * 16 + nn;
#pragma unroll
  for (int j = 0; j < 8; ++j) {
    const size_t r0 = (size_t)(tM0 * 16 + j + 8 * half) * N;
    const size_t r1 = (size_t)(tM0 * 16 + 16 + j + 8 * half) * N;
    C[r0 + n0]      = a00[j];
    C[r0 + n0 + 16] = a01[j];
    C[r1 + n0]      = a10[j];
    C[r1 + n0 + 16] = a11[j];
  }
}

// ---------------- rank contraction + RoPE ----------------
// one thread per (b,s,h): builds qh[B,H,S,DK] (scaled by 1/(QR*DK)),
// kh[B,H,S,DK] (1/RK), vT[B,H,DK,S] (1/RK), all bf16.
__global__ void k_rank_rope(const float* __restrict__ Aq, const float* __restrict__ Ak,
                            const float* __restrict__ Av, const float* __restrict__ Bq,
                            const float* __restrict__ Bk, const float* __restrict__ Bv,
                            __bf16* __restrict__ qh, __bf16* __restrict__ kh,
                            __bf16* __restrict__ vT) {
  int tid = blockIdx.x * blockDim.x + threadIdx.x;
  if (tid >= Bn * Sn * Hn) return;
  const int h = tid % Hn;
  const int s = (tid / Hn) % Sn;
  const int b = tid / (Hn * Sn);
  const size_t bs = (size_t)b * Sn + s;

  float aq[QR], ak[RK], av[RK];
#pragma unroll
  for (int r = 0; r < QR; ++r) aq[r] = Aq[bs * NAq + h * QR + r];
#pragma unroll
  for (int r = 0; r < RK; ++r) {
    ak[r] = Ak[bs * NAk + h * RK + r];
    av[r] = Av[bs * NAv + h * RK + r];
  }
  const float* bq = Bq + bs * NBq;  // [QR][DK]
  const float* bk = Bk + bs * NBk;  // [RK][DK]
  const float* bv = Bv + bs * NBv;  // [RK][DK]

  const size_t qo = ((size_t)(b * Hn + h) * Sn + s) * DK;
  const size_t vo = (size_t)(b * Hn + h) * DK * Sn;

  const float qscale = 1.0f / (QR * DK);  // folds the 1/DK attention scale
  const float kscale = 1.0f / RK;
  const float lnbase = 9.210340371976184f;  // ln(10000)
  for (int i = 0; i < 32; ++i) {
    float invf = __expf(-(2.0f * i / 64.0f) * lnbase);
    float sn, cs;
    __sincosf((float)s * invf, &sn, &cs);

    float q1 = 0.f, q2 = 0.f;
#pragma unroll
    for (int r = 0; r < QR; ++r) {
      float x1 = bq[r * DK + i], x2 = bq[r * DK + i + 32];
      q1 += aq[r] * ( x1 * cs + x2 * sn);
      q2 += aq[r] * (-x1 * sn + x2 * cs);
    }
    qh[qo + i]      = (__bf16)(q1 * qscale);
    qh[qo + i + 32] = (__bf16)(q2 * qscale);

    float k1 = 0.f, k2 = 0.f;
#pragma unroll
    for (int r = 0; r < RK; ++r) {
      float x1 = bk[r * DK + i], x2 = bk[r * DK + i + 32];
      k1 += ak[r] * ( x1 * cs + x2 * sn);
      k2 += ak[r] * (-x1 * sn + x2 * cs);
    }
    kh[qo + i]      = (__bf16)(k1 * kscale);
    kh[qo + i + 32] = (__bf16)(k2 * kscale);

    float v1 = 0.f, v2 = 0.f;
#pragma unroll
    for (int r = 0; r < RK; ++r) {
      v1 += av[r] * bv[r * DK + i];
      v2 += av[r] * bv[r * DK + i + 32];
    }
    vT[vo + (size_t)i * Sn + s]        = (__bf16)(v1 * kscale);
    vT[vo + (size_t)(i + 32) * Sn + s] = (__bf16)(v2 * kscale);
  }
}

// ---------------- causal flash attention, one wave per 16-query tile ----------------
// qh,kh: [B,H,S,DK] bf16 ; vT: [B,H,DK,S] bf16 ; ho: [B,S,H*DK] bf16
__global__ void __launch_bounds__(256)
k_flash(const __bf16* __restrict__ qh, const __bf16* __restrict__ kh,
        const __bf16* __restrict__ vT, __bf16* __restrict__ ho) {
  __shared__ __bf16 pbuf[8][16 * 32];  // wave-private P staging (1KB/wave)

  const int lane = threadIdx.x & 31;
  const int wid  = threadIdx.x >> 5;
  const int qtile = blockIdx.x * 8 + wid;  // 0..127
  const int bh = blockIdx.y;               // 0..31
  const int b = bh >> 4;
  const int h = bh & 15;
  const int qbase = qtile * 16;
  const int nn   = lane & 15;
  const int half = lane >> 4;
  const float NEG_INF = -__builtin_inff();

  // Q fragments for both 32-wide d halves
  const __bf16* qrow = qh + ((size_t)bh * Sn + qbase + nn) * DK;
  const v16bf qf0 = load_a_frag(qrow, half);       // d 0..31
  const v16bf qf1 = load_a_frag(qrow + 32, half);  // d 32..63

  float mrun[8], lrun[8];
  v8f acc[4];
#pragma unroll
  for (int j = 0; j < 8; ++j) { mrun[j] = NEG_INF; lrun[j] = 0.f; }
#pragma unroll
  for (int t = 0; t < 4; ++t) acc[t] = v8f{};

  const int kend = qbase + 16;  // exclusive key bound (causal)
  for (int kb = 0; kb < kend; kb += 32) {
    // ---- scores for two 16-key subtiles (1/DK pre-folded into qh) ----
    v8f sc[2];
#pragma unroll
    for (int t = 0; t < 2; ++t) {
      const int key0 = kb + t * 16;
      const __bf16* krow = kh + ((size_t)bh * Sn + key0 + nn) * DK;
      v16bf kf0 = *(const v16bf*)(krow + half * 16);        // d = half*16+e
      v16bf kf1 = *(const v16bf*)(krow + 32 + half * 16);   // d = 32+half*16+e
      v8f s = {};
      s = WMMA_BF16(qf0, kf0, s);
      s = WMMA_BF16(qf1, kf1, s);
      const int key = key0 + nn;
#pragma unroll
      for (int j = 0; j < 8; ++j) {
        const int row = qbase + j + 8 * half;
        s[j] = (key <= row) ? s[j] : NEG_INF;
      }
      sc[t] = s;
    }

    // ---- online softmax update (rows split across 16-lane halves) ----
    float mnew[8], alpha[8];
#pragma unroll
    for (int j = 0; j < 8; ++j) {
      float r = fmaxf(sc[0][j], sc[1][j]);
      for (int off = 1; off < 16; off <<= 1)
        r = fmaxf(r, __shfl_xor(r, off, 32));
      mnew[j]  = fmaxf(mrun[j], r);
      alpha[j] = (mrun[j] == NEG_INF) ? 0.f : __expf(mrun[j] - mnew[j]);
    }
#pragma unroll
    for (int j = 0; j < 8; ++j) {
      float p0 = __expf(sc[0][j] - mnew[j]);
      float p1 = __expf(sc[1][j] - mnew[j]);
      const int row16 = j + 8 * half;
      pbuf[wid][row16 * 32 + nn]      = (__bf16)p0;
      pbuf[wid][row16 * 32 + 16 + nn] = (__bf16)p1;
      float r = p0 + p1;
      for (int off = 1; off < 16; off <<= 1)
        r += __shfl_xor(r, off, 32);
      lrun[j] = lrun[j] * alpha[j] + r;
      mrun[j] = mnew[j];
#pragma unroll
      for (int t = 0; t < 4; ++t) acc[t][j] *= alpha[j];
    }

    // ---- P (C-layout) -> A-fragment via wave-private LDS (DS ops in-order) ----
    const __bf16* prow = &pbuf[wid][nn * 32];
    v16bf pf = load_a_frag(prow, half);

    // ---- PV: 4 d-tiles of 16, K = 32 keys ----
#pragma unroll
    for (int dt = 0; dt < 4; ++dt) {
      const __bf16* vrow = vT + ((size_t)bh * DK + dt * 16 + nn) * Sn + kb + half * 16;
      v16bf vf = *(const v16bf*)vrow;
      acc[dt] = WMMA_BF16(pf, vf, acc[dt]);
    }
  }

  // ---- epilogue: normalize, store ho[b, row, h*64 + d] ----
  float inv[8];
#pragma unroll
  for (int j = 0; j < 8; ++j) inv[j] = 1.0f / lrun[j];
#pragma unroll
  for (int dt = 0; dt < 4; ++dt) {
#pragma unroll
    for (int j = 0; j < 8; ++j) {
      const int row = qbase + j + 8 * half;
      const int col = h * DK + dt * 16 + nn;
      ho[((size_t)b * Sn + row) * Dm + col] = (__bf16)(acc[dt][j] * inv[j]);
    }
  }
}

// ---------------- host side ----------------
static inline dim3 gemm_grid(int M, int N) {
  return dim3((N + 127) / 128, (M + 63) / 64, 1);
}

extern "C" void kernel_launch(void* const* d_in, const int* in_sizes, int n_in,
                              void* d_out, int out_size, void* d_ws, size_t ws_size,
                              hipStream_t stream) {
  (void)in_sizes; (void)n_in; (void)out_size; (void)ws_size;
  const float* q    = (const float*)d_in[0];
  const float* k    = (const float*)d_in[1];
  const float* v    = (const float*)d_in[2];
  // d_in[3] = mask (bool) — causal, handled analytically
  const float* W_Aq = (const float*)d_in[4];
  const float* W_Ak = (const float*)d_in[5];
  const float* W_Av = (const float*)d_in[6];
  const float* W_Bq = (const float*)d_in[7];
  const float* W_Bk = (const float*)d_in[8];
  const float* W_Bv = (const float*)d_in[9];
  const float* Wo   = (const float*)d_in[10];
  float* out = (float*)d_out;

  char* wsp = (char*)d_ws;
  auto alloc = [&](size_t bytes) -> void* {
    void* r = (void*)wsp;
    wsp += (bytes + 255) & ~(size_t)255;
    return r;
  };
  const size_t BS = (size_t)Bn * Sn;

  __bf16* qb  = (__bf16*)alloc(BS * Dm * 2);
  __bf16* kb_ = (__bf16*)alloc(BS * Dm * 2);
  __bf16* vb  = (__bf16*)alloc(BS * Dm * 2);
  __bf16* wAq = (__bf16*)alloc((size_t)Dm * NAq * 2);   // transposed [N,K]
  __bf16* wAk = (__bf16*)alloc((size_t)Dm * NAk * 2);
  __bf16* wAv = (__bf16*)alloc((size_t)Dm * NAv * 2);
  __bf16* wBq = (__bf16*)alloc((size_t)Dm * NBq * 2);
  __bf16* wBk = (__bf16*)alloc((size_t)Dm * NBk * 2);
  __bf16* wBv = (__bf16*)alloc((size_t)Dm * NBv * 2);
  __bf16* wO  = (__bf16*)alloc((size_t)Dm * Dm * 2);
  float* Aq = (float*)alloc(BS * NAq * 4);
  float* Ak = (float*)alloc(BS * NAk * 4);
  float* Av = (float*)alloc(BS * NAv * 4);
  float* Bq = (float*)alloc(BS * NBq * 4);
  float* Bk = (float*)alloc(BS * NBk * 4);
  float* Bv = (float*)alloc(BS * NBv * 4);
  __bf16* qh = (__bf16*)alloc((size_t)Bn * Hn * Sn * DK * 2);
  __bf16* kh = (__bf16*)alloc((size_t)Bn * Hn * Sn * DK * 2);
  __bf16* vT = (__bf16*)alloc((size_t)Bn * Hn * DK * Sn * 2);
  __bf16* ho = (__bf16*)alloc(BS * Dm * 2);

  auto cvt = [&](const float* s, __bf16* d, size_t n) {
    k_cvt_bf16<<<(unsigned)((n + 255) / 256), 256, 0, stream>>>(s, d, (int)n);
  };
  auto cvt_t = [&](const float* s, __bf16* d, int K, int N) {
    size_t n = (size_t)K * N;
    k_cvt_t_bf16<<<(unsigned)((n + 255) / 256), 256, 0, stream>>>(s, d, K, N);
  };
  cvt(q, qb, BS * Dm);
  cvt(k, kb_, BS * Dm);
  cvt(v, vb, BS * Dm);
  cvt_t(W_Aq, wAq, Dm, NAq);
  cvt_t(W_Ak, wAk, Dm, NAk);
  cvt_t(W_Av, wAv, Dm, NAv);
  cvt_t(W_Bq, wBq, Dm, NBq);
  cvt_t(W_Bk, wBk, Dm, NBk);
  cvt_t(W_Bv, wBv, Dm, NBv);
  cvt_t(Wo,  wO,  Dm, Dm);

  const int M = (int)BS;  // 4096
  k_gemm_bf16<<<gemm_grid(M, NAq), 256, 0, stream>>>(qb,  wAq, Aq, M, NAq, Dm);
  k_gemm_bf16<<<gemm_grid(M, NAk), 256, 0, stream>>>(kb_, wAk, Ak, M, NAk, Dm);
  k_gemm_bf16<<<gemm_grid(M, NAv), 256, 0, stream>>>(vb,  wAv, Av, M, NAv, Dm);
  k_gemm_bf16<<<gemm_grid(M, NBq), 256, 0, stream>>>(qb,  wBq, Bq, M, NBq, Dm);
  k_gemm_bf16<<<gemm_grid(M, NBk), 256, 0, stream>>>(kb_, wBk, Bk, M, NBk, Dm);
  k_gemm_bf16<<<gemm_grid(M, NBv), 256, 0, stream>>>(vb,  wBv, Bv, M, NBv, Dm);

  const int nrr = Bn * Sn * Hn;
  k_rank_rope<<<(nrr + 255) / 256, 256, 0, stream>>>(Aq, Ak, Av, Bq, Bk, Bv, qh, kh, vT);

  k_flash<<<dim3(Sn / 16 / 8, Bn * Hn, 1), 256, 0, stream>>>(qh, kh, vT, ho);

  k_gemm_bf16<<<gemm_grid(M, Dm), 256, 0, stream>>>(ho, wO, out, M, Dm, Dm);
}
